// TransformerEncoder_9766755631496
// MI455X (gfx1250) — compile-verified, matrix operand order checked by reading, not run on hardware
//
#include <hip/hip_runtime.h>

// ---------------------------------------------------------------------------
// Transformer encoder for MI455X (gfx1250, wave32, WMMA).
// B=4, L=1024, SD=64, D=512, H=8, E=64, FF=2048, NL=4.
// All matmuls run through v_wmma_f32_16x16x32_f16 (f16 in, f32 accumulate).
// ---------------------------------------------------------------------------

typedef __attribute__((ext_vector_type(16))) _Float16 v16h;
typedef __attribute__((ext_vector_type(8)))  float    v8f;
typedef __attribute__((ext_vector_type(4)))  unsigned int v4u;

#define DEV __device__ __forceinline__

// ---------------------------------------------------------------------------
// Fragment loader: 16x32 f16 tile slice for one lane (A layout, ISA 7.12.2).
// Lane (group g = lane>>4) holds K pairs {0..7}+8g and {16..23}+8g, i.e. two
// contiguous 4-dword runs -> two ds_load_b128.  rowbase must be 16B aligned.
// The same loader serves B fragments when the tile is stored "column-major"
// (row index = output column, contiguous K).
// ---------------------------------------------------------------------------
DEV v16h ld_frag(const _Float16* rowbase, int g) {
    union { v4u u[2]; v16h h; } r;
    const v4u* p = (const v4u*)rowbase;
    r.u[0] = p[g];       // dwords  g*4 .. g*4+3     (K = 8g+0..7)
    r.u[1] = p[2 + g];   // dwords  8+g*4 .. 8+g*4+3 (K = 16+8g+0..7)
    return r.h;
}

// ---------------------------------------------------------------------------
// GEMM: C = epilogue(A(MxK,f32) @ W(KxN,f32) + bias)
// Block: 256 threads (8 waves), tile 128(M) x 128(N), K-chunk 32.
// Wave grid 4x2; each wave owns a 32x64 tile = 2x4 WMMA tiles
// (8 WMMAs per K-chunk against 6 fragment fetches -> high MMA/LDS ratio).
// Register prefetch overlaps next chunk's global loads with current WMMAs.
// ACT: 0=none, 1=LeakyReLU(0.01), 2=GeLU(exact, erf)
// QKV: scatter N=1536 columns to Q/K/V buffers laid out [B,H,L,E]
//      (reference column index = j*512 + e*8 + h, heads fastest).
// ---------------------------------------------------------------------------
constexpr int GBM = 128, GBN = 128, GBK = 32;
constexpr int GLDA = 40;   // halves per LDS row (32 data + 8 pad; 80B, 16B-aligned)

template<int ACT, bool BIAS, bool RES, bool QKV>
__global__ __launch_bounds__(256) void gemm_wmma_kernel(
    const float* __restrict__ A, const float* __restrict__ W,
    const float* __restrict__ bias, const float* __restrict__ res,
    float* __restrict__ C, int M, int N, int K,
    float* __restrict__ qb, float* __restrict__ kb, float* __restrict__ vb)
{
    __shared__ __align__(16) _Float16 Al[GBM * GLDA];
    __shared__ __align__(16) _Float16 Wl[GBN * GLDA];

    const int t    = threadIdx.x;
    const int lane = t & 31;
    const int w    = t >> 5;        // wave 0..7
    const int wm   = w >> 1;        // 0..3 (M)
    const int wn   = w & 1;         // 0..1 (N)
    const int g    = lane >> 4;     // half-wave group
    const int lr   = lane & 15;
    const int bm   = blockIdx.y * GBM;
    const int bn   = blockIdx.x * GBN;

    // staging assignments
    const int ar = t >> 1;          // A row 0..127
    const int ac = (t & 1) << 4;    // A col 0 / 16
    const int sn = t & 127;         // W col 0..127
    const int k0 = (t >> 7) << 4;   // W k 0 / 16

    float4 aReg[4];
    float  wReg[16];

    auto load_chunk = [&](int kk) {
        const float4* Ap = (const float4*)(A + (size_t)(bm + ar) * K + kk + ac);
        #pragma unroll
        for (int i = 0; i < 4; ++i) aReg[i] = Ap[i];
        const float* Wp = W + (size_t)(kk + k0) * N + bn + sn;
        #pragma unroll
        for (int i = 0; i < 16; ++i) wReg[i] = Wp[(size_t)i * N];
    };
    auto store_chunk = [&]() {
        _Float16* da = &Al[ar * GLDA + ac];
        #pragma unroll
        for (int i = 0; i < 4; ++i) {
            da[i * 4 + 0] = (_Float16)aReg[i].x;
            da[i * 4 + 1] = (_Float16)aReg[i].y;
            da[i * 4 + 2] = (_Float16)aReg[i].z;
            da[i * 4 + 3] = (_Float16)aReg[i].w;
        }
        _Float16* dw = &Wl[sn * GLDA + k0];   // transposed: Wt[n][k]
        #pragma unroll
        for (int i = 0; i < 16; ++i) dw[i] = (_Float16)wReg[i];
    };

    v8f acc[2][4] = {};

    load_chunk(0);
    store_chunk();
    __syncthreads();

    for (int kk = 0; kk < K; kk += GBK) {
        const bool more = (kk + GBK) < K;   // uniform across block
        if (more) load_chunk(kk + GBK);     // prefetch while WMMAs run

        v16h af[2], bf[4];
        #pragma unroll
        for (int mi = 0; mi < 2; ++mi)
            af[mi] = ld_frag(&Al[(wm * 32 + mi * 16 + lr) * GLDA], g);
        #pragma unroll
        for (int ni = 0; ni < 4; ++ni)
            bf[ni] = ld_frag(&Wl[(wn * 64 + ni * 16 + lr) * GLDA], g);
        #pragma unroll
        for (int mi = 0; mi < 2; ++mi)
            #pragma unroll
            for (int ni = 0; ni < 4; ++ni)
                acc[mi][ni] = __builtin_amdgcn_wmma_f32_16x16x32_f16(
                    false, af[mi], false, bf[ni], (short)0, acc[mi][ni],
                    false, false);

        if (more) {
            __syncthreads();     // all waves done reading current tiles
            store_chunk();
            __syncthreads();     // next tiles visible
        }
    }

    // ---- epilogue ----
    #pragma unroll
    for (int mi = 0; mi < 2; ++mi) {
        #pragma unroll
        for (int ni = 0; ni < 4; ++ni) {
            #pragma unroll
            for (int r = 0; r < 8; ++r) {
                const int gm = bm + wm * 32 + mi * 16 + r + g * 8;
                const int gn = bn + wn * 64 + ni * 16 + lr;
                float val = acc[mi][ni][r];
                if constexpr (BIAS) val += bias[gn];
                if constexpr (ACT == 1) val = val >= 0.f ? val : 0.01f * val;
                if constexpr (ACT == 2)
                    val = 0.5f * val * (1.f + erff(val * 0.70710678118654752f));
                if constexpr (RES) val += res[(size_t)gm * N + gn];
                if constexpr (QKV) {
                    const int j  = gn >> 9;          // 0=q 1=k 2=v
                    const int rr = gn & 511;
                    const int e  = rr >> 3;          // head dim (cols are e*8+h)
                    const int h  = rr & 7;
                    const int b  = gm >> 10;
                    const int l  = gm & 1023;
                    float* dst = (j == 0) ? qb : (j == 1) ? kb : vb;
                    dst[(((size_t)(b * 8 + h)) * 1024 + l) * 64 + e] = val;
                } else {
                    C[(size_t)gm * N + gn] = val;
                }
            }
        }
    }
}

// ---------------------------------------------------------------------------
// LayerNorm over D=512, one block (128 threads) per row; optional pos-emb add.
// ---------------------------------------------------------------------------
__global__ __launch_bounds__(128) void ln_kernel(
    const float* __restrict__ in, float* __restrict__ out,
    const float* __restrict__ gamma, const float* __restrict__ beta,
    const float* __restrict__ pos)
{
    const int row = blockIdx.x;          // 0..4095
    const int t   = threadIdx.x;         // 0..127
    const float4 xv = ((const float4*)(in + (size_t)row * 512))[t];

    float s  = xv.x + xv.y + xv.z + xv.w;
    float s2 = xv.x * xv.x + xv.y * xv.y + xv.z * xv.z + xv.w * xv.w;
    #pragma unroll
    for (int m = 1; m < 32; m <<= 1) {
        s  += __shfl_xor(s,  m, 32);
        s2 += __shfl_xor(s2, m, 32);
    }
    __shared__ float sb[4], sb2[4];
    if ((t & 31) == 0) { sb[t >> 5] = s; sb2[t >> 5] = s2; }
    __syncthreads();
    s  = sb[0]  + sb[1]  + sb[2]  + sb[3];
    s2 = sb2[0] + sb2[1] + sb2[2] + sb2[3];

    const float mean = s * (1.f / 512.f);
    const float var  = s2 * (1.f / 512.f) - mean * mean;
    const float rs   = rsqrtf(var + 1e-5f);

    const int c = t * 4;
    float4 o;
    o.x = (xv.x - mean) * rs * gamma[c + 0] + beta[c + 0];
    o.y = (xv.y - mean) * rs * gamma[c + 1] + beta[c + 1];
    o.z = (xv.z - mean) * rs * gamma[c + 2] + beta[c + 2];
    o.w = (xv.w - mean) * rs * gamma[c + 3] + beta[c + 3];
    if (pos) {
        const float4 pv = ((const float4*)(pos + (size_t)(row & 1023) * 512))[t];
        o.x += pv.x; o.y += pv.y; o.z += pv.z; o.w += pv.w;
    }
    ((float4*)(out + (size_t)row * 512))[t] = o;
}

// ---------------------------------------------------------------------------
// Flash attention (causal).  Q/K/V: [B*H, L, 64] f32.  O: [B, L, H*64] f32.
// Block = 128 threads (4 waves) handles one (b,h) x 64 query rows; each wave
// owns 16 rows.  Key loop in 64-wide tiles with online softmax.
// ---------------------------------------------------------------------------
constexpr int ALD = 72;   // halves per LDS row (64 data + 8 pad; 144B aligned)

__global__ __launch_bounds__(128) void attn_kernel(
    const float* __restrict__ Q, const float* __restrict__ Kc,
    const float* __restrict__ V, float* __restrict__ O)
{
    __shared__ __align__(16) _Float16 Ql[64 * ALD];
    __shared__ __align__(16) _Float16 Kl[64 * ALD];
    __shared__ __align__(16) _Float16 Vt[64 * ALD];       // Vt[e][n]
    __shared__ __align__(16) _Float16 Pl[4 * 16 * ALD];   // per-wave P tiles

    const int t    = threadIdx.x;
    const int lane = t & 31;
    const int w    = t >> 5;       // wave 0..3 -> query rows w*16..w*16+15
    const int g    = lane >> 4;
    const int lr   = lane & 15;
    const int bh   = blockIdx.y;   // 0..31
    const int qt   = blockIdx.x;   // 0..15
    const int qb   = qt * 64;
    const size_t base = (size_t)bh * 1024 * 64;

    // stage Q tile (64x64) once
    {
        const int n  = t >> 1;
        const int e0 = (t & 1) * 32;
        const float4* qp = (const float4*)(Q + base + (size_t)(qb + n) * 64 + e0);
        _Float16* dst = &Ql[n * ALD + e0];
        #pragma unroll
        for (int i = 0; i < 8; ++i) {
            float4 f = qp[i];
            dst[i * 4 + 0] = (_Float16)f.x;
            dst[i * 4 + 1] = (_Float16)f.y;
            dst[i * 4 + 2] = (_Float16)f.z;
            dst[i * 4 + 3] = (_Float16)f.w;
        }
    }

    float mrow[8], lrow[8];
    #pragma unroll
    for (int r = 0; r < 8; ++r) { mrow[r] = -__builtin_inff(); lrow[r] = 0.f; }
    v8f oacc[4] = {};

    _Float16* pw = &Pl[w * 16 * ALD];

    for (int jt = 0; jt <= qt; ++jt) {
        __syncthreads();   // previous iteration's K/V reads done
        // stage K tile row-major + V tile transposed
        {
            const int n  = t >> 1;
            const int e0 = (t & 1) * 32;
            const float4* kp = (const float4*)(Kc + base + (size_t)(jt * 64 + n) * 64 + e0);
            const float4* vp = (const float4*)(V  + base + (size_t)(jt * 64 + n) * 64 + e0);
            _Float16* kd = &Kl[n * ALD + e0];
            #pragma unroll
            for (int i = 0; i < 8; ++i) {
                float4 f = kp[i];
                kd[i * 4 + 0] = (_Float16)f.x;
                kd[i * 4 + 1] = (_Float16)f.y;
                kd[i * 4 + 2] = (_Float16)f.z;
                kd[i * 4 + 3] = (_Float16)f.w;
                float4 fv = vp[i];
                Vt[(e0 + i * 4 + 0) * ALD + n] = (_Float16)fv.x;
                Vt[(e0 + i * 4 + 1) * ALD + n] = (_Float16)fv.y;
                Vt[(e0 + i * 4 + 2) * ALD + n] = (_Float16)fv.z;
                Vt[(e0 + i * 4 + 3) * ALD + n] = (_Float16)fv.w;
            }
        }
        __syncthreads();

        // S = Q(16x64) @ K^T(64x64)  -> 4 tiles of 16x16
        v8f sacc[4] = {};
        #pragma unroll
        for (int kc = 0; kc < 64; kc += 32) {
            v16h a = ld_frag(&Ql[(w * 16 + lr) * ALD + kc], g);
            #pragma unroll
            for (int nt = 0; nt < 4; ++nt) {
                v16h b = ld_frag(&Kl[(nt * 16 + lr) * ALD + kc], g);
                sacc[nt] = __builtin_amdgcn_wmma_f32_16x16x32_f16(
                    false, a, false, b, (short)0, sacc[nt], false, false);
            }
        }

        // scale + causal mask + row max
        float sv[4][8];
        float rm[8];
        #pragma unroll
        for (int r = 0; r < 8; ++r) rm[r] = -__builtin_inff();
        #pragma unroll
        for (int nt = 0; nt < 4; ++nt) {
            const int col = jt * 64 + nt * 16 + lr;
            #pragma unroll
            for (int r = 0; r < 8; ++r) {
                const int rowi = qb + w * 16 + r + g * 8;
                float x = sacc[nt][r] * 0.125f;        // 1/sqrt(64)
                if (col > rowi) x = -__builtin_inff();
                sv[nt][r] = x;
                rm[r] = fmaxf(rm[r], x);
            }
        }
        #pragma unroll
        for (int r = 0; r < 8; ++r)
            #pragma unroll
            for (int m = 1; m < 16; m <<= 1)
                rm[r] = fmaxf(rm[r], __shfl_xor(rm[r], m, 32));

        // online softmax update
        float al[8], rs[8];
        #pragma unroll
        for (int r = 0; r < 8; ++r) {
            const float mn = fmaxf(mrow[r], rm[r]);
            al[r] = __expf(mrow[r] - mn);   // exp(-inf)=0 on first tile
            mrow[r] = mn;
            rs[r] = 0.f;
        }
        #pragma unroll
        for (int nt = 0; nt < 4; ++nt)
            #pragma unroll
            for (int r = 0; r < 8; ++r) {
                const float p = __expf(sv[nt][r] - mrow[r]);
                rs[r] += p;
                pw[(r + g * 8) * ALD + nt * 16 + lr] = (_Float16)p;
            }
        #pragma unroll
        for (int r = 0; r < 8; ++r) {
            #pragma unroll
            for (int m = 1; m < 16; m <<= 1) rs[r] += __shfl_xor(rs[r], m, 32);
            lrow[r] = lrow[r] * al[r] + rs[r];
        }
        #pragma unroll
        for (int et = 0; et < 4; ++et)
            #pragma unroll
            for (int r = 0; r < 8; ++r)
                oacc[et][r] *= al[r];

        __syncthreads();   // make P tiles visible / order LDS writes

        // O += P(16x64) @ V(64x64)   (Vt[e][n] gives B-layout fragments)
        #pragma unroll
        for (int kc = 0; kc < 64; kc += 32) {
            v16h a = ld_frag(&pw[lr * ALD + kc], g);
            #pragma unroll
            for (int et = 0; et < 4; ++et) {
                v16h b = ld_frag(&Vt[(et * 16 + lr) * ALD + kc], g);
                oacc[et] = __builtin_amdgcn_wmma_f32_16x16x32_f16(
                    false, a, false, b, (short)0, oacc[et], false, false);
            }
        }
    }

    // write O as [B, L, H*64]  (reference reshape: o[..., h*E + e])
    const int b = bh >> 3, h = bh & 7;
    #pragma unroll
    for (int et = 0; et < 4; ++et)
        #pragma unroll
        for (int r = 0; r < 8; ++r) {
            const int rowi = qb + w * 16 + r + g * 8;
            const int e = et * 16 + lr;
            O[((size_t)(b * 1024 + rowi)) * 512 + h * 64 + e] =
                oacc[et][r] * (1.f / lrow[r]);
        }
}

// ---------------------------------------------------------------------------
// Host launcher
// ---------------------------------------------------------------------------
extern "C" void kernel_launch(void* const* d_in, const int* in_sizes, int n_in,
                              void* d_out, int out_size, void* d_ws, size_t ws_size,
                              hipStream_t stream) {
    (void)in_sizes; (void)n_in; (void)out_size; (void)ws_size;

    const float* states   = (const float*)d_in[0];
    /* d_in[1] = mask (exact causal; applied analytically) */
    const float* pos_emb  = (const float*)d_in[2];
    const float* emb_W1   = (const float*)d_in[3];
    const float* emb_b1   = (const float*)d_in[4];
    const float* emb_W2   = (const float*)d_in[5];
    const float* emb_b2   = (const float*)d_in[6];
    const float* emb_ln_g = (const float*)d_in[7];
    const float* emb_ln_b = (const float*)d_in[8];
    const float* Wqkv     = (const float*)d_in[9];
    const float* Wout     = (const float*)d_in[10];
    const float* bout     = (const float*)d_in[11];
    const float* ln1_g    = (const float*)d_in[12];
    const float* ln1_b    = (const float*)d_in[13];
    const float* ln3_g    = (const float*)d_in[14];
    const float* ln3_b    = (const float*)d_in[15];
    const float* ffW1     = (const float*)d_in[16];
    const float* ffb1     = (const float*)d_in[17];
    const float* ffW2     = (const float*)d_in[18];
    const float* ffb2     = (const float*)d_in[19];
    const float* fin_g    = (const float*)d_in[20];
    const float* fin_b    = (const float*)d_in[21];

    // workspace layout (48 MB): x | y | Q K V O  (FFN hidden overlays Q..O)
    char* ws = (char*)d_ws;
    const size_t MB = 1u << 20;
    float* x    = (float*)(ws + 0 * MB);     // 4096x512
    float* y    = (float*)(ws + 8 * MB);     // 4096x512
    float* qbuf = (float*)(ws + 16 * MB);    // 32x1024x64
    float* kbuf = (float*)(ws + 24 * MB);
    float* vbuf = (float*)(ws + 32 * MB);
    float* obuf = (float*)(ws + 40 * MB);    // 4096x512
    float* fbuf = (float*)(ws + 16 * MB);    // 4096x2048 (reuses q/k/v/o)

    const dim3 blk(256);
    const dim3 g512 (512  / GBN, 4096 / GBM);
    const dim3 g1536(1536 / GBN, 4096 / GBM);
    const dim3 g2048(2048 / GBN, 4096 / GBM);
    const dim3 lnb(128);

    // obs embedding: Linear->LeakyReLU->Linear->LN(+pos)
    gemm_wmma_kernel<1, true, false, false><<<g512, blk, 0, stream>>>(
        states, emb_W1, emb_b1, nullptr, y, 4096, 512, 64,
        nullptr, nullptr, nullptr);
    gemm_wmma_kernel<0, true, false, false><<<g512, blk, 0, stream>>>(
        y, emb_W2, emb_b2, nullptr, x, 4096, 512, 512,
        nullptr, nullptr, nullptr);
    ln_kernel<<<4096, lnb, 0, stream>>>(x, x, emb_ln_g, emb_ln_b, pos_emb);

    for (int i = 0; i < 4; ++i) {
        // pre-norm attention
        ln_kernel<<<4096, lnb, 0, stream>>>(x, y, ln1_g + i * 512,
                                            ln1_b + i * 512, nullptr);
        gemm_wmma_kernel<0, false, false, true><<<g1536, blk, 0, stream>>>(
            y, Wqkv + (size_t)i * 512 * 1536, nullptr, nullptr, nullptr,
            4096, 1536, 512, qbuf, kbuf, vbuf);
        attn_kernel<<<dim3(16, 32), lnb, 0, stream>>>(qbuf, kbuf, vbuf, obuf);
        gemm_wmma_kernel<0, true, true, false><<<g512, blk, 0, stream>>>(
            obuf, Wout + (size_t)i * 512 * 512, bout + i * 512, x, x,
            4096, 512, 512, nullptr, nullptr, nullptr);
        // pre-norm FFN
        ln_kernel<<<4096, lnb, 0, stream>>>(x, y, ln3_g + i * 512,
                                            ln3_b + i * 512, nullptr);
        gemm_wmma_kernel<2, true, false, false><<<g2048, blk, 0, stream>>>(
            y, ffW1 + (size_t)i * 512 * 2048, ffb1 + i * 2048, nullptr, fbuf,
            4096, 2048, 512, nullptr, nullptr, nullptr);
        gemm_wmma_kernel<0, true, true, false><<<g512, blk, 0, stream>>>(
            fbuf, ffW2 + (size_t)i * 2048 * 512, ffb2 + i * 512, x, x,
            4096, 512, 2048, nullptr, nullptr, nullptr);
    }

    ln_kernel<<<4096, lnb, 0, stream>>>(x, (float*)d_out, fin_g, fin_b, nullptr);
}
